// HeteroEdgePredictor_per_node_25950192403098
// MI455X (gfx1250) — matrix operand order repaired and checked by reading, not compiled, not run on hardware
//
#include <hip/hip_runtime.h>

typedef __attribute__((ext_vector_type(16))) _Float16 v16h;
typedef __attribute__((ext_vector_type(8)))  _Float16 v8h;
typedef __attribute__((ext_vector_type(8)))  float    v8f;

#define TT 8
#define DD 512
#define HH 100
#define CC 2
#define NTILE 7          // ceil(100/16) column tiles (H padded to 112)
#define KSTEPS 16        // 512 / 32
#define GEMM_WAVES 7
#define GEMM_THREADS (GEMM_WAVES * 32)
#define APAD 520         // 512 + 8 halves row padding (LDS bank spread)
#define EPAD 113         // 112 + 1 f32 padding

// ---------------- stable counting sort by edge type ----------------

__global__ void k_hist(const int* __restrict__ types, int E, int* __restrict__ blockCounts) {
    __shared__ int cnt[TT];
    int tid = threadIdx.x;
    if (tid < TT) cnt[tid] = 0;
    __syncthreads();
    int i = blockIdx.x * 256 + tid;
    if (i < E) atomicAdd(&cnt[types[i]], 1);
    __syncthreads();
    if (tid < TT) blockCounts[blockIdx.x * TT + tid] = cnt[tid];
}

__global__ void k_scan(const int* __restrict__ blockCounts, int nblk,
                       int* __restrict__ blockBase, int* __restrict__ typeStart,
                       int* __restrict__ tileInfo, int* __restrict__ numTiles) {
    __shared__ int totals[TT];
    int tid = threadIdx.x;
    if (tid < TT) {
        int run = 0;
        for (int b = 0; b < nblk; ++b) {
            blockBase[b * TT + tid] = run;
            run += blockCounts[b * TT + tid];
        }
        totals[tid] = run;
    }
    __syncthreads();
    if (tid == 0) {
        int run = 0;
        for (int t = 0; t < TT; ++t) { typeStart[t] = run; run += totals[t]; }
        typeStart[TT] = run;
    }
    __syncthreads();
    if (tid < TT) {
        int base = typeStart[tid];
        for (int b = 0; b < nblk; ++b) blockBase[b * TT + tid] += base;
    }
    __syncthreads();
    if (tid == 0) {
        int nt = 0;
        for (int t = 0; t < TT; ++t) {
            int cnt = totals[t];
            int start = typeStart[t];
            for (int j = 0; j * 16 < cnt; ++j) {
                tileInfo[2 * nt]     = t;
                tileInfo[2 * nt + 1] = start + j * 16;
                ++nt;
            }
        }
        numTiles[0] = nt;
    }
}

__global__ void k_scatter(const int* __restrict__ types, int E,
                          const int* __restrict__ blockBase, int* __restrict__ perm) {
    __shared__ int st[256];
    int tid = threadIdx.x;
    int i = blockIdx.x * 256 + tid;
    int t = (i < E) ? types[i] : -1;
    st[tid] = t;
    __syncthreads();
    if (i < E) {
        int rank = 0;
        for (int j = 0; j < tid; ++j) rank += (st[j] == t);
        perm[blockBase[blockIdx.x * TT + t] + rank] = i;
    }
}

// ------- repack src_W/dst_W (f32, T x D x H) into f16 WMMA B-fragment order -------
// slot = ((((mat*T + t)*NTILE + nt)*KSTEPS + ks)*32 + lane)*8 + v  (each slot = 2 halves)
// B layout (16-bit, 32x16): lane n = lane&15; K = ks*32 + (lane>=16?16:0) + 2v + j

__global__ void k_w16(const float* __restrict__ srcW, const float* __restrict__ dstW,
                      _Float16* __restrict__ w16, int nslots) {
    int idx = blockIdx.x * 256 + threadIdx.x;
    if (idx >= nslots) return;
    int v    = idx & 7;
    int r1   = idx >> 3;
    int lane = r1 & 31;
    int r2   = r1 >> 5;
    int ks   = r2 & 15;
    int r3   = r2 >> 4;
    int nt   = r3 % NTILE;
    int r4   = r3 / NTILE;
    int t    = r4 & 7;
    int mat  = r4 >> 3;
    int k0 = ks * 32 + ((lane >> 4) << 4) + v * 2;
    int n  = nt * 16 + (lane & 15);
    const float* W = mat ? dstW : srcW;
    float a = 0.f, b = 0.f;
    if (n < HH) {
        a = W[((size_t)t * DD + k0) * HH + n];
        b = W[((size_t)t * DD + k0 + 1) * HH + n];
    }
    w16[(size_t)idx * 2]     = (_Float16)a;
    w16[(size_t)idx * 2 + 1] = (_Float16)b;
}

// ---------------- h_save copy (biggest byte mover) ----------------

__global__ void k_copy(const float4* __restrict__ src, float4* __restrict__ dst, long n4) {
    long i = (long)blockIdx.x * blockDim.x + threadIdx.x;
    long stride = (long)gridDim.x * blockDim.x;
    for (; i < n4; i += stride) dst[i] = src[i];
}

// ---------------- main per-tile WMMA GEMM ----------------

__global__ __launch_bounds__(GEMM_THREADS)
void k_gemm(const float* __restrict__ h,
            const float* __restrict__ srcB, const float* __restrict__ dstB,
            const float* __restrict__ outW, const float* __restrict__ outB,
            const int* __restrict__ perm, const int* __restrict__ typeStart,
            const int* __restrict__ tileInfo, const int* __restrict__ numTiles,
            const _Float16* __restrict__ w16, int E,
            float* __restrict__ posOut, float* __restrict__ negOut) {
    if ((int)blockIdx.x >= numTiles[0]) return;   // uniform per block

    __shared__ __align__(16) _Float16 As[3][16][APAD];  // src/pos/neg A rows, f16
    __shared__ int   rowE[16];
    __shared__ float Eg[2][16][EPAD];                   // pos/neg edge features

    int tid = threadIdx.x;
    int t    = tileInfo[2 * blockIdx.x];
    int i0   = tileInfo[2 * blockIdx.x + 1];
    int tEnd = typeStart[t + 1];

    if (tid < 16) {
        int i = i0 + tid;
        rowE[tid] = perm[(i < tEnd) ? i : i0];
    }
    __syncthreads();

    // gather 3 x 16 rows of h (f32 -> f16) into LDS, coalesced along k
    for (int idx = tid; idx < 3 * 16 * DD; idx += GEMM_THREADS) {
        int s   = idx >> 13;       // / (16*512)
        int rem = idx & 8191;
        int r   = rem >> 9;
        int k   = rem & 511;
        int e   = rowE[r];
        As[s][r][k] = (_Float16)h[((size_t)s * E + e) * DD + k];
    }
    __syncthreads();

    int lane = tid & 31;
    int wv   = tid >> 5;           // wave = column tile 0..6
    int m16  = lane & 15;
    int lh   = lane >> 4;

    v8f accS = {}; v8f accP = {}; v8f accN = {};
    size_t fs = ((size_t)(0 * TT + t) * NTILE + wv) * KSTEPS;
    size_t fd = ((size_t)(1 * TT + t) * NTILE + wv) * KSTEPS;
    const v16h* Wf = (const v16h*)w16;

    #pragma unroll 2
    for (int ks = 0; ks < KSTEPS; ++ks) {
        int kb = ks * 32 + lh * 16;
        // A 16x32 f16 layout: elements [0..15] = K kb..kb+15
        v8h s0 = *(const v8h*)&As[0][m16][kb];
        v8h s1 = *(const v8h*)&As[0][m16][kb + 8];
        v8h p0 = *(const v8h*)&As[1][m16][kb];
        v8h p1 = *(const v8h*)&As[1][m16][kb + 8];
        v8h n0 = *(const v8h*)&As[2][m16][kb];
        v8h n1 = *(const v8h*)&As[2][m16][kb + 8];
        v16h aS = __builtin_shufflevector(s0, s1, 0,1,2,3,4,5,6,7,8,9,10,11,12,13,14,15);
        v16h aP = __builtin_shufflevector(p0, p1, 0,1,2,3,4,5,6,7,8,9,10,11,12,13,14,15);
        v16h aN = __builtin_shufflevector(n0, n1, 0,1,2,3,4,5,6,7,8,9,10,11,12,13,14,15);
        v16h bS = Wf[(fs + ks) * 32 + lane];
        v16h bD = Wf[(fd + ks) * 32 + lane];
        accS = __builtin_amdgcn_wmma_f32_16x16x32_f16(false, aS, false, bS, (short)0, accS, false, false);
        accP = __builtin_amdgcn_wmma_f32_16x16x32_f16(false, aP, false, bD, (short)0, accP, false, false);
        accN = __builtin_amdgcn_wmma_f32_16x16x32_f16(false, aN, false, bD, (short)0, accN, false, false);
    }

    // epilogue: bias + ReLU; C/D layout: row M = v + lh*8, col n = wv*16 + m16
    int n = wv * 16 + m16;
    float bias = 0.f;
    if (n < HH) bias = srcB[t * HH + n] + dstB[t * HH + n];
    #pragma unroll
    for (int v = 0; v < 8; ++v) {
        int M = v + lh * 8;
        float s  = accS[v];
        float pe = s + accP[v] + bias;
        float ne = s + accN[v] + bias;
        Eg[0][M][n] = fmaxf(pe, 0.f);
        Eg[1][M][n] = fmaxf(ne, 0.f);
    }
    __syncthreads();

    // tiny H(100) -> C(2) head: 64 lanes, one (edge, class, pos/neg) each
    if (tid < 64) {
        int e  = tid & 15;
        int c  = (tid >> 4) & 1;
        int pn = tid >> 5;
        float sum = outB[t * CC + c];
        const float* Wo = outW + (size_t)t * HH * CC + c;
        for (int hh = 0; hh < HH; ++hh)
            sum += Eg[pn][e][hh] * Wo[hh * CC];
        int i = i0 + e;
        if (i < tEnd) {
            float* dst = pn ? negOut : posOut;
            dst[(size_t)i * CC + c] = sum;
        }
    }
}

// ---------------- launch ----------------

extern "C" void kernel_launch(void* const* d_in, const int* in_sizes, int n_in,
                              void* d_out, int out_size, void* d_ws, size_t ws_size,
                              hipStream_t stream) {
    const float* h    = (const float*)d_in[0];
    const float* srcW = (const float*)d_in[1];
    const float* srcB = (const float*)d_in[2];
    const float* dstW = (const float*)d_in[3];
    const float* dstB = (const float*)d_in[4];
    const float* outW = (const float*)d_in[5];
    const float* outB = (const float*)d_in[6];
    const int*   types = (const int*)d_in[7];
    int E = in_sizes[7];
    int nblk = (E + 255) / 256;
    int maxTiles = E / 16 + TT;

    char* ws = (char*)d_ws;
    size_t off = 0;
    auto alloc = [&](size_t bytes) { size_t cur = off; off = (off + bytes + 255) & ~(size_t)255; return cur; };
    int* perm        = (int*)(ws + alloc((size_t)E * 4));
    int* blockCounts = (int*)(ws + alloc((size_t)nblk * TT * 4));
    int* blockBase   = (int*)(ws + alloc((size_t)nblk * TT * 4));
    int* tStart      = (int*)(ws + alloc((TT + 1) * 4));
    int* tileInfo    = (int*)(ws + alloc((size_t)maxTiles * 2 * 4));
    int* numTiles    = (int*)(ws + alloc(4));
    int nslots = 2 * TT * NTILE * KSTEPS * 32 * 8;          // u32 slots of packed f16 pairs
    _Float16* w16 = (_Float16*)(ws + alloc((size_t)nslots * 4));

    float* posOut = (float*)d_out;
    float* negOut = posOut + (size_t)E * CC;
    float* hSave  = negOut + (size_t)E * CC;

    k_hist<<<nblk, 256, 0, stream>>>(types, E, blockCounts);
    k_scan<<<1, 256, 0, stream>>>(blockCounts, nblk, blockBase, tStart, tileInfo, numTiles);
    k_scatter<<<nblk, 256, 0, stream>>>(types, E, blockBase, perm);
    k_w16<<<(nslots + 255) / 256, 256, 0, stream>>>(srcW, dstW, w16, nslots);
    long n4 = (long)2 * E * DD / 4;
    k_copy<<<4096, 256, 0, stream>>>((const float4*)h, (float4*)hSave, n4);
    k_gemm<<<maxTiles, GEMM_THREADS, 0, stream>>>(h, srcB, dstB, outW, outB,
                                                  perm, tStart, tileInfo, numTiles,
                                                  w16, E, posOut, negOut);
}
